// DGCNN_repsurf_81466939670830
// MI455X (gfx1250) — matrix operand order repaired and checked by reading, not compiled, not run on hardware
//
#include <hip/hip_runtime.h>
#include <hip/hip_bf16.h>

// ---------------------------------------------------------------------------
// DGCNN-RepSurf forward for MI455X (gfx1250, wave32, WMMA).
// Heavy GEMMs (q/k/v/d projections, q·k^T, attn·v) run on
// v_wmma_f32_16x16x32_bf16 (bf16 inputs, f32 accumulate), 64x64 output tile
// per wave (4x4 WMMA tiles, 32 FLOP/byte register-level reuse). Everything
// else (knn top-k, umbrella geometry, layernorm, softmax, edge convs, head)
// is fp32 VALU work.
//
// Input flattening assumption: d_in[0] = x, then params flattened as a JAX
// pytree (dict keys sorted alphabetically, recursively) -> 106 tensors total.
// ---------------------------------------------------------------------------

#define BB    2
#define NN    1000
#define NPTS_ 10000
#define KT    50
#define KSEL  20
#define KP    1024
#define FINF  3.402823466e38f

typedef unsigned short u16;
typedef unsigned int   u32;
typedef __attribute__((ext_vector_type(16))) __bf16 v16bf;
typedef __attribute__((ext_vector_type(8)))  float  v8f;

struct __align__(16) Q16 { u32 x, y, z, w; };
union FragU { Q16 q[2]; v16bf v; };

__device__ __forceinline__ u16 f2bf(float f) {
  u32 u = __float_as_uint(f);
  u32 r = u + 0x7FFFu + ((u >> 16) & 1u);   // round-to-nearest-even
  return (u16)(r >> 16);
}
__device__ __forceinline__ float bf2f(u16 h) {
  return __uint_as_float(((u32)h) << 16);
}

// A-fragment: 16x32 bf16, row-major source (ld = K stride).
// Lane L<16: M=L, K in {k0..k0+7, k0+16..k0+23}; lane L>=16: M=L-16, K +8.
__device__ __forceinline__ v16bf load_frag_a(const u16* base, int ld, int row0,
                                             int k0, int lane) {
  int row = row0 + (lane & 15);
  int kb  = k0 + ((lane & 16) ? 8 : 0);
  const u16* p = base + (size_t)row * ld + kb;
  FragU u;
  u.q[0] = *reinterpret_cast<const Q16*>(p);        // K kb..kb+7
  u.q[1] = *reinterpret_cast<const Q16*>(p + 16);   // K kb+16..kb+23
  return u.v;
}
// B-fragment: 32x16 bf16 from N-major (row = output col, contiguous K).
// Lane L<16: N=L, K in {k0..k0+7, k0+16..k0+23}; lane L>=16: N=L-16, K +8.
__device__ __forceinline__ v16bf load_frag_b(const u16* base, int ld, int col0,
                                             int k0, int lane) {
  int col = col0 + (lane & 15);
  int kb  = k0 + ((lane & 16) ? 8 : 0);
  const u16* p = base + (size_t)col * ld + kb;
  FragU u;
  u.q[0] = *reinterpret_cast<const Q16*>(p);
  u.q[1] = *reinterpret_cast<const Q16*>(p + 16);
  return u.v;
}

// C[m, j] = alpha * sum_k A[m,k]*B[j,k] + bias[j].
// One wave per 64x64 output region (4x4 WMMA tiles). K multiple of 32.
// Both padded dims (Mt4*64, Nt4*64) must be covered by the padded buffers.
__global__ void __launch_bounds__(128)
gemm_nt_bf16(const u16* __restrict__ A, int lda,
             const u16* __restrict__ Bm, int ldb,
             int Mt4, int Nt4, int K,
             const float* __restrict__ bias, float alpha,
             float* __restrict__ Cf, u16* __restrict__ Cb,
             int ldc, int Mstore, int Nstore, int transC) {
  int wave = blockIdx.x * 4 + (threadIdx.x >> 5);
  int lane = threadIdx.x & 31;
  if (wave >= Mt4 * Nt4) return;                 // wave-uniform exit (EXEC all-1)
  int tm = wave / Nt4, tn = wave % Nt4;
  int m0 = tm * 64, n0 = tn * 64;
  v8f acc[4][4];
  {
    v8f z = {0.f, 0.f, 0.f, 0.f, 0.f, 0.f, 0.f, 0.f};
#pragma unroll
    for (int i = 0; i < 4; i++)
#pragma unroll
      for (int j = 0; j < 4; j++) acc[i][j] = z;
  }
  for (int k0 = 0; k0 < K; k0 += 32) {
    v16bf a[4], b[4];
#pragma unroll
    for (int i = 0; i < 4; i++) a[i] = load_frag_a(A, lda, m0 + 16 * i, k0, lane);
#pragma unroll
    for (int j = 0; j < 4; j++) b[j] = load_frag_b(Bm, ldb, n0 + 16 * j, k0, lane);
#pragma unroll
    for (int i = 0; i < 4; i++)
#pragma unroll
      for (int j = 0; j < 4; j++)
        acc[i][j] = __builtin_amdgcn_wmma_f32_16x16x32_bf16(
            false, a[i], false, b[j], (short)0, acc[i][j], false, false);
  }
  int cl = lane & 15;
  int rh = (lane & 16) ? 8 : 0;
#pragma unroll
  for (int i = 0; i < 4; i++) {
#pragma unroll
    for (int j = 0; j < 4; j++) {
      int mb  = m0 + 16 * i;
      int col = n0 + 16 * j + cl;
      float badd = (bias != nullptr && col < Nstore) ? bias[col] : 0.f;
#pragma unroll
      for (int jj = 0; jj < 8; jj++) {
        int row = mb + rh + jj;
        if (row < Mstore && col < Nstore) {
          float v = acc[i][j][jj] * alpha + badd;
          size_t off = transC ? ((size_t)col * ldc + row) : ((size_t)row * ldc + col);
          if (Cf) Cf[off] = v;
          if (Cb) Cb[off] = f2bf(v);
        }
      }
    }
  }
}

// f32 -> bf16, zero-padded to (rowsPad x ldd)
__global__ void conv2bf16(const float* __restrict__ src, u16* __restrict__ dst,
                          int rows, int cols, int ldd, int rowsPad) {
  int c = blockIdx.x * blockDim.x + threadIdx.x;
  int r = blockIdx.y;
  if (c >= ldd || r >= rowsPad) return;
  float v = (r < rows && c < cols) ? src[(size_t)r * cols + c] : 0.f;
  dst[(size_t)r * ldd + c] = f2bf(v);
}

// gather selected points: pts (B,N,3) and ptsT (B,3,N)
__global__ void gather_kernel(const float* __restrict__ x, float* __restrict__ pts,
                              float* __restrict__ ptsT) {
  int t = blockIdx.x * blockDim.x + threadIdx.x;
  if (t >= BB * NN) return;
  int b = t / NN, n = t % NN;
  const int S = NPTS_ + NN;
  const float* xb = x + (size_t)b * 3 * S;
  int idx = (int)xb[NPTS_ + n];          // channel 0, slot 10000+n
  float px = xb[idx];
  float py = xb[S + idx];
  float pz = xb[2 * S + idx];
  pts[(size_t)t * 3 + 0] = px;
  pts[(size_t)t * 3 + 1] = py;
  pts[(size_t)t * 3 + 2] = pz;
  float* pT = ptsT + (size_t)b * 3 * NN;
  pT[n] = px; pT[NN + n] = py; pT[2 * NN + n] = pz;
}

__global__ void sqnorm_kernel(const float* __restrict__ xin, float* __restrict__ xsq, int C) {
  int t = blockIdx.x * blockDim.x + threadIdx.x;
  if (t >= BB * NN) return;
  int b = t / NN, n = t % NN;
  const float* xb = xin + (size_t)b * C * NN;
  float s = 0.f;
  for (int c = 0; c < C; c++) { float v = xb[(size_t)c * NN + n]; s += v * v; }
  xsq[t] = s;
}

// dist[b,n,m] = 2*x_n.x_m - |x_n|^2 - |x_m|^2   (top_k of this = knn)
__global__ void dist_kernel(const float* __restrict__ xin, const float* __restrict__ xsq,
                            float* __restrict__ dist, int C) {
  int m = blockIdx.x * blockDim.x + threadIdx.x;
  int n = blockIdx.y, b = blockIdx.z;
  if (m >= NN) return;
  const float* xb = xin + (size_t)b * C * NN;
  float s = 0.f;
  for (int c = 0; c < C; c++) s += xb[(size_t)c * NN + n] * xb[(size_t)c * NN + m];
  dist[((size_t)b * NN + n) * NN + m] = 2.f * s - xsq[b * NN + n] - xsq[b * NN + m];
}

// stable descending top-k via lexicographic scan-selection (matches lax.top_k)
__global__ void topk_desc(const float* __restrict__ dist, int* __restrict__ out, int k) {
  int t = blockIdx.x * blockDim.x + threadIdx.x;
  if (t >= BB * NN) return;
  const float* row = dist + (size_t)t * NN;
  int* o = out + (size_t)t * k;
  float lastV = FINF; int lastI = -1;
  for (int j = 0; j < k; j++) {
    float bv = -FINF; int bi = 0; bool found = false;
    for (int m = 0; m < NN; m++) {
      float v = row[m];
      bool after = (v < lastV) || (v == lastV && m > lastI);
      if (after && (!found || v > bv)) { bv = v; bi = m; found = true; }
    }
    o[j] = bi; lastV = bv; lastI = bi;
  }
}

// umbrella geometry + tiny 7->7->7->7 MLP + max over k=9 -> h0 (B,10,N)
__global__ void umbrella_kernel(const float* __restrict__ pts, const int* __restrict__ idx10,
    const float* __restrict__ w1, const float* __restrict__ w2, const float* __restrict__ w3,
    const float* __restrict__ b2, const float* __restrict__ b3,
    const float* bn1b, const float* bn1g, const float* bn1rm, const float* bn1rv,
    const float* bn2b, const float* bn2g, const float* bn2rm, const float* bn2rv,
    float* __restrict__ h0) {
  int t = blockIdx.x * blockDim.x + threadIdx.x;
  if (t >= BB * NN) return;
  int b = t / NN, n = t % NN;
  const float* P = pts + (size_t)b * NN * 3;
  float cx = P[n * 3], cy = P[n * 3 + 1], cz = P[n * 3 + 2];
  float xn[9][3]; float phi[9];
  const float inv2pi = 0.15915494309189535f;
  for (int j = 0; j < 9; j++) {
    int nb = idx10[((size_t)b * NN + n) * 10 + 1 + j];   // drop self (slot 0)
    xn[j][0] = P[nb * 3]     - cx;
    xn[j][1] = P[nb * 3 + 1] - cy;
    xn[j][2] = P[nb * 3 + 2] - cz;
    phi[j] = atan2f(xn[j][1], xn[j][0]) * inv2pi + 0.5f;
  }
  int ord[9];
  for (int j = 0; j < 9; j++) ord[j] = j;
  for (int i = 1; i < 9; i++) {                 // stable ascending insertion sort
    int oi = ord[i]; float pv = phi[oi]; int j = i - 1;
    while (j >= 0 && phi[ord[j]] > pv) { ord[j + 1] = ord[j]; j--; }
    ord[j + 1] = oi;
  }
  float cen[9][3], nrm[9][3];
  for (int k = 0; k < 9; k++) {
    const float* v1 = xn[ord[k]];
    const float* v2 = xn[ord[(k + 1) % 9]];
    cen[k][0] = 0.5f * (v1[0] + v2[0]);
    cen[k][1] = 0.5f * (v1[1] + v2[1]);
    cen[k][2] = 0.5f * (v1[2] + v2[2]);
    float nx = v1[1] * v2[2] - v1[2] * v2[1] + 1e-5f;
    float ny = v1[2] * v2[0] - v1[0] * v2[2] + 1e-5f;
    float nz = v1[0] * v2[1] - v1[1] * v2[0] + 1e-5f;
    float nn = sqrtf(nx * nx + ny * ny + nz * nz);
    nrm[k][0] = nx / nn; nrm[k][1] = ny / nn; nrm[k][2] = nz / nn;
  }
  float mask = (nrm[0][0] > 0.f) ? 1.f : -1.f;
  const float inv3 = 0.5773502691896258f;  // 1/sqrt(3)
  float fmx[7];
  for (int o = 0; o < 7; o++) fmx[o] = -FINF;
  for (int k = 0; k < 9; k++) {
    float f[7], t1[7], t2[7];
    f[0] = cen[k][0]; f[1] = cen[k][1]; f[2] = cen[k][2];
    f[3] = nrm[k][0] * mask; f[4] = nrm[k][1] * mask; f[5] = nrm[k][2] * mask;
    f[6] = (cen[k][0] * f[3] + cen[k][1] * f[4] + cen[k][2] * f[5]) * inv3;
    for (int o = 0; o < 7; o++) {
      float s = 0.f;
      for (int i = 0; i < 7; i++) s += w1[o * 7 + i] * f[i];
      s = (s - bn1rm[o]) * rsqrtf(bn1rv[o] + 1e-5f) * bn1g[o] + bn1b[o];
      t1[o] = fmaxf(s, 0.f);
    }
    for (int o = 0; o < 7; o++) {
      float s = b2[o];
      for (int i = 0; i < 7; i++) s += w2[o * 7 + i] * t1[i];
      s = (s - bn2rm[o]) * rsqrtf(bn2rv[o] + 1e-5f) * bn2g[o] + bn2b[o];
      t2[o] = fmaxf(s, 0.f);
    }
    for (int o = 0; o < 7; o++) {
      float s = b3[o];
      for (int i = 0; i < 7; i++) s += w3[o * 7 + i] * t2[i];
      fmx[o] = fmaxf(fmx[o], s);
    }
  }
  float* H = h0 + (size_t)b * 10 * NN;
  H[n] = cx; H[NN + n] = cy; H[2 * NN + n] = cz;
  for (int o = 0; o < 7; o++) H[(size_t)(3 + o) * NN + n] = fmx[o];
}

// xv[m,n] = xin[c, idx50[n,kk]] - xin[c,n], m = kk*C + c ; writes f32 + padded bf16
__global__ void build_xv(const float* __restrict__ xin, const int* __restrict__ idx50,
                         float* __restrict__ xvf, u16* __restrict__ xvb,
                         int C, int M, int Mp) {
  int n = blockIdx.x * blockDim.x + threadIdx.x;   // 0..KP-1
  int m = blockIdx.y;
  if (n >= KP || m >= Mp) return;
  float v = 0.f;
  if (m < M && n < NN) {
    int kk = m / C, c = m % C;
    int nb = idx50[(size_t)n * KT + kk];
    v = xin[(size_t)c * NN + nb] - xin[(size_t)c * NN + n];
  }
  if (n < NN) xvf[(size_t)m * NN + n] = v;
  xvb[(size_t)m * KP + n] = f2bf(v);
}

// in-place row softmax of bf16 logits, zero-pad cols [M,Mp)
__global__ void softmax_rows(u16* __restrict__ attn, int M, int Mp) {
  int m = blockIdx.x;
  u16* row = attn + (size_t)m * Mp;
  __shared__ float red[256];
  int tid = threadIdx.x;
  float mx = -FINF;
  for (int j = tid; j < M; j += 256) mx = fmaxf(mx, bf2f(row[j]));
  red[tid] = mx; __syncthreads();
  for (int s = 128; s > 0; s >>= 1) { if (tid < s) red[tid] = fmaxf(red[tid], red[tid + s]); __syncthreads(); }
  mx = red[0]; __syncthreads();
  float sum = 0.f;
  for (int j = tid; j < M; j += 256) sum += expf(bf2f(row[j]) - mx);
  red[tid] = sum; __syncthreads();
  for (int s = 128; s > 0; s >>= 1) { if (tid < s) red[tid] += red[tid + s]; __syncthreads(); }
  float inv = 1.f / red[0]; __syncthreads();
  for (int j = tid; j < M; j += 256) row[j] = f2bf(expf(bf2f(row[j]) - mx) * inv);
  for (int j = M + tid; j < Mp; j += 256) row[j] = 0;
}

// out = LayerNorm(xa + xb) * g + b over last dim Kn; optional padded bf16 mirror
__global__ void ln_rows(const float* __restrict__ xa, const float* __restrict__ xb,
                        const float* __restrict__ g, const float* __restrict__ bb,
                        float* __restrict__ outf, u16* outb, int Kn, int ldb2) {
  int m = blockIdx.x;
  const float* ra = xa + (size_t)m * Kn;
  const float* rb = xb + (size_t)m * Kn;
  __shared__ float s1[256], s2[256];
  int tid = threadIdx.x;
  float su = 0.f, sq = 0.f;
  for (int j = tid; j < Kn; j += 256) { float v = ra[j] + rb[j]; su += v; sq += v * v; }
  s1[tid] = su; s2[tid] = sq; __syncthreads();
  for (int st = 128; st > 0; st >>= 1) {
    if (tid < st) { s1[tid] += s1[tid + st]; s2[tid] += s2[tid + st]; }
    __syncthreads();
  }
  float mu  = s1[0] / (float)Kn;
  float var = s2[0] / (float)Kn - mu * mu;
  float inv = rsqrtf(var + 1e-5f);
  float* wo = outf + (size_t)m * Kn;
  u16* wb = outb ? outb + (size_t)m * ldb2 : nullptr;
  for (int j = tid; j < Kn; j += 256) {
    float v = ra[j] + rb[j];
    float y = (v - mu) * inv * g[j] + bb[j];
    wo[j] = y;
    if (wb) wb[j] = f2bf(y);
  }
  if (wb) for (int j = Kn + tid; j < ldb2; j += 256) wb[j] = 0;
}

// scores[n,kk] = sum_c (h3+xv)[kk*C+c, n] * wfc[c] + bfc
__global__ void scores_kernel(const float* __restrict__ h3, const float* __restrict__ xvf,
                              const float* __restrict__ wfc, const float* __restrict__ bfc,
                              float* __restrict__ sc, int C) {
  int n = blockIdx.x * blockDim.x + threadIdx.x;
  int kk = blockIdx.y;
  if (n >= NN) return;
  float s = bfc[0];
  size_t basem = (size_t)kk * C;
  for (int c = 0; c < C; c++) {
    size_t r = (basem + c) * NN + n;
    s += (h3[r] + xvf[r]) * wfc[c];
  }
  sc[(size_t)n * KT + kk] = s;
}

// stable ascending argsort, keep first 20 of 50 (matches jnp.argsort[:k])
__global__ void order20_kernel(const float* __restrict__ sc, int* __restrict__ ord) {
  int n = blockIdx.x * blockDim.x + threadIdx.x;
  if (n >= NN) return;
  const float* row = sc + (size_t)n * KT;
  float lastV = -FINF; int lastI = -1;
  for (int j = 0; j < KSEL; j++) {
    float bv = FINF; int bi = 0; bool found = false;
    for (int k = 0; k < KT; k++) {
      float v = row[k];
      bool after = (v > lastV) || (v == lastV && k > lastI);
      if (after && (!found || v < bv)) { bv = v; bi = k; found = true; }
    }
    ord[(size_t)n * KSEL + j] = bi; lastV = bv; lastI = bi;
  }
}

// edge conv: max_kk relu(bn( sum_c w[o,c]*(rel - x) + w[o,C+c]*x ))
__global__ void edgeconv_kernel(const float* __restrict__ xin, const float* __restrict__ xvf,
    const int* __restrict__ ord, const float* __restrict__ w,
    const float* g, const float* bb, const float* rm, const float* rv,
    float* __restrict__ xout, int C, int O) {
  int n = blockIdx.x * blockDim.x + threadIdx.x;
  int o = blockIdx.y;
  if (n >= NN) return;
  const float* wo = w + (size_t)o * 2 * C;
  float ctr = 0.f;
  for (int c = 0; c < C; c++) ctr += wo[C + c] * xin[(size_t)c * NN + n];
  float scale = rsqrtf(rv[o] + 1e-5f) * g[o];
  float best = -FINF;
  for (int kk = 0; kk < KSEL; kk++) {
    int sk = ord[(size_t)n * KSEL + kk];
    float s = ctr;
    for (int c = 0; c < C; c++) {
      float rel = xvf[((size_t)sk * C + c) * NN + n];
      s += wo[c] * (rel - xin[(size_t)c * NN + n]);
    }
    s = fmaxf((s - rm[o]) * scale + bb[o], 0.f);
    best = fmaxf(best, s);
  }
  xout[(size_t)o * NN + n] = best;
}

// ec5 (1024x512) + bn + relu + max over N
__global__ void ec5max_kernel(const float* x1, const float* x2, const float* x3, const float* x4,
    const float* __restrict__ w, const float* g, const float* bb,
    const float* rm, const float* rv, float* __restrict__ out) {
  int o = blockIdx.x, b = blockIdx.y;
  const float* wo = w + (size_t)o * 512;
  const float* p1 = x1 + (size_t)b * 64 * NN;
  const float* p2 = x2 + (size_t)b * 64 * NN;
  const float* p3 = x3 + (size_t)b * 128 * NN;
  const float* p4 = x4 + (size_t)b * 256 * NN;
  float scale = rsqrtf(rv[o] + 1e-5f) * g[o];
  float best = -FINF;
  for (int n = threadIdx.x; n < NN; n += 256) {
    float s = 0.f;
    for (int i = 0; i < 64; i++)  s += wo[i]       * p1[(size_t)i * NN + n];
    for (int i = 0; i < 64; i++)  s += wo[64 + i]  * p2[(size_t)i * NN + n];
    for (int i = 0; i < 128; i++) s += wo[128 + i] * p3[(size_t)i * NN + n];
    for (int i = 0; i < 256; i++) s += wo[256 + i] * p4[(size_t)i * NN + n];
    s = fmaxf((s - rm[o]) * scale + bb[o], 0.f);
    best = fmaxf(best, s);
  }
  __shared__ float red[256];
  red[threadIdx.x] = best; __syncthreads();
  for (int st = 128; st > 0; st >>= 1) {
    if (threadIdx.x < st) red[threadIdx.x] = fmaxf(red[threadIdx.x], red[threadIdx.x + st]);
    __syncthreads();
  }
  if (threadIdx.x == 0) out[(size_t)b * 1024 + o] = red[0];
}

__global__ void dense_kernel(const float* __restrict__ in, const float* __restrict__ w,
    const float* g, const float* bb, const float* rm, const float* rv,
    float* __restrict__ out, int I, int O, int bnrelu) {
  int o = blockIdx.x * blockDim.x + threadIdx.x;
  int b = blockIdx.y;
  if (o >= O) return;
  const float* wo = w + (size_t)o * I;
  const float* x  = in + (size_t)b * I;
  float s = 0.f;
  for (int i = 0; i < I; i++) s += wo[i] * x[i];
  if (bnrelu) s = fmaxf((s - rm[o]) * rsqrtf(rv[o] + 1e-5f) * g[o] + bb[o], 0.f);
  out[(size_t)b * O + o] = s;
}

// ---------------------------------------------------------------------------
extern "C" void kernel_launch(void* const* d_in, const int* in_sizes, int n_in,
                              void* d_out, int out_size, void* d_ws, size_t ws_size,
                              hipStream_t stream) {
  (void)in_sizes; (void)out_size; (void)ws_size;
  if (n_in < 106) return;   // param-order assumption violated; nothing safe to do
  auto F = [&](int i) -> const float* { return (const float*)d_in[i]; };

  // Flat input indices (x first, params pytree-sorted):
  // 1-4 bn6 b/g/rm/rv, 5-8 bn7, 9-12 ec1_bn, 13 ec1_w, 14-17 ec2_bn, 18 ec2_w,
  // 19-22 ec3_bn, 23 ec3_w, 24-27 ec4_bn, 28 ec4_w, 29-32 ec5_bn, 33 ec5_w,
  // 34.. knn1 {bd,bfc,bk,bq,bv,ln1_b,ln1_g,ln2_b,ln2_g,wd,wfc,wk,wq,wv},
  // 48 knn2, 62 knn3, 76 knn4, 90 mlp1, 91 mlp2, 92 mlp3,
  // 93 umb.b2, 94 umb.b3, 95-98 umb.bn1, 99-102 umb.bn2, 103-105 umb.w1/2/3

  // ---- workspace arena (~284 MB) ----
  char* wsbase = (char*)d_ws; size_t wsoff = 0;
  auto alloc = [&](size_t bytes) -> void* {
    wsoff = (wsoff + 255) & ~(size_t)255;
    void* p = wsbase + wsoff; wsoff += bytes; return p;
  };
  const int MAXMP = KT * 128;   // 6400 (block4)

  float* pts   = (float*)alloc((size_t)BB * NN * 3 * 4);
  float* ptsT  = (float*)alloc((size_t)BB * 3 * NN * 4);
  float* h0    = (float*)alloc((size_t)BB * 10 * NN * 4);
  float* x1    = (float*)alloc((size_t)BB * 64 * NN * 4);
  float* x2    = (float*)alloc((size_t)BB * 64 * NN * 4);
  float* x3    = (float*)alloc((size_t)BB * 128 * NN * 4);
  float* x4    = (float*)alloc((size_t)BB * 256 * NN * 4);
  float* feat  = (float*)alloc((size_t)BB * 1024 * 4);
  float* fcA   = (float*)alloc((size_t)BB * 512 * 4);
  float* fcB   = (float*)alloc((size_t)BB * 256 * 4);
  float* xsq   = (float*)alloc((size_t)BB * NN * 4);
  float* dist  = (float*)alloc((size_t)BB * NN * NN * 4);
  int*   idx10 = (int*)alloc((size_t)BB * NN * 10 * 4);
  int*   idx50 = (int*)alloc((size_t)BB * NN * KT * 4);
  float* scb   = (float*)alloc((size_t)NN * KT * 4);
  int*   ordb  = (int*)alloc((size_t)NN * KSEL * 4);
  u16* wq_b = (u16*)alloc((size_t)KP * KP * 2);
  u16* wk_b = (u16*)alloc((size_t)KP * KP * 2);
  u16* wv_b = (u16*)alloc((size_t)KP * KP * 2);
  u16* wd_b = (u16*)alloc((size_t)KP * KP * 2);
  float* xv_f   = (float*)alloc((size_t)MAXMP * NN * 4);
  u16*   xv_b   = (u16*)alloc((size_t)MAXMP * KP * 2);
  u16*   q_b    = (u16*)alloc((size_t)MAXMP * KP * 2);
  u16*   k_b    = (u16*)alloc((size_t)MAXMP * KP * 2);
  u16*   vT_b   = (u16*)alloc((size_t)KP * MAXMP * 2);
  u16*   attn_b = (u16*)alloc((size_t)MAXMP * MAXMP * 2);
  float* out_f  = (float*)alloc((size_t)MAXMP * NN * 4);
  float* out1_f = (float*)alloc((size_t)MAXMP * NN * 4);
  u16*   out1_b = (u16*)alloc((size_t)MAXMP * KP * 2);
  float* h2_f   = (float*)alloc((size_t)MAXMP * NN * 4);
  float* h3_f   = (float*)alloc((size_t)MAXMP * NN * 4);

  // Mp / Np here are always multiples of 64 (512, 1024, 3200, 6400).
  auto gemm = [&](const u16* A, int lda, const u16* Bm, int ldb,
                  int Mp, int Np64, int K, const float* bias, float alpha,
                  float* Cf, u16* Cb, int ldc, int Mst, int Nst, int tr) {
    int Mt4 = Mp / 64, Nt4 = Np64 / 64;
    int wgs = (Mt4 * Nt4 + 3) / 4;
    gemm_nt_bf16<<<dim3(wgs), dim3(128), 0, stream>>>(
        A, lda, Bm, ldb, Mt4, Nt4, K, bias, alpha, Cf, Cb, ldc, Mst, Nst, tr);
  };

  // ================= stage 1: gather + umbrella =================
  gather_kernel<<<dim3((BB * NN + 255) / 256), dim3(256), 0, stream>>>(F(0), pts, ptsT);
  sqnorm_kernel<<<dim3((BB * NN + 255) / 256), dim3(256), 0, stream>>>(ptsT, xsq, 3);
  dist_kernel<<<dim3((NN + 255) / 256, NN, BB), dim3(256), 0, stream>>>(ptsT, xsq, dist, 3);
  topk_desc<<<dim3((BB * NN + 127) / 128), dim3(128), 0, stream>>>(dist, idx10, 10);
  umbrella_kernel<<<dim3((BB * NN + 63) / 64), dim3(64), 0, stream>>>(
      pts, idx10, F(103), F(104), F(105), F(93), F(94),
      F(95), F(96), F(97), F(98), F(99), F(100), F(101), F(102), h0);

  // ================= stage 2: edge blocks =================
  auto edge_block = [&](const float* xin, int C, int O, int kb, const float* convw,
                        const float* bnb, const float* bng, const float* bnrm,
                        const float* bnrv, float* xout) {
    int M = KT * C;
    int Mp = (M + 63) & ~63;            // multiple of 64 (512, 3200, 6400)
    // weights -> bf16, zero-padded 1024x1024 (wq=kb+12, wk=kb+11, wv=kb+13, wd=kb+9)
    conv2bf16<<<dim3(KP / 256, KP), dim3(256), 0, stream>>>(F(kb + 12), wq_b, 1000, 1000, KP, KP);
    conv2bf16<<<dim3(KP / 256, KP), dim3(256), 0, stream>>>(F(kb + 11), wk_b, 1000, 1000, KP, KP);
    conv2bf16<<<dim3(KP / 256, KP), dim3(256), 0, stream>>>(F(kb + 13), wv_b, 1000, 1000, KP, KP);
    conv2bf16<<<dim3(KP / 256, KP), dim3(256), 0, stream>>>(F(kb + 9),  wd_b, 1000, 1000, KP, KP);
    // knn (both batches)
    sqnorm_kernel<<<dim3((BB * NN + 255) / 256), dim3(256), 0, stream>>>(xin, xsq, C);
    dist_kernel<<<dim3((NN + 255) / 256, NN, BB), dim3(256), 0, stream>>>(xin, xsq, dist, C);
    topk_desc<<<dim3((BB * NN + 127) / 128), dim3(128), 0, stream>>>(dist, idx50, KT);
    for (int b = 0; b < BB; b++) {
      const float* xb = xin + (size_t)b * C * NN;
      const int* idb = idx50 + (size_t)b * NN * KT;
      hipMemsetAsync(vT_b, 0, (size_t)KP * Mp * 2, stream);
      build_xv<<<dim3(KP / 256, Mp), dim3(256), 0, stream>>>(xb, idb, xv_f, xv_b, C, M, Mp);
      // projections (WMMA bf16)
      gemm(xv_b, KP, wq_b, KP, Mp, KP, KP, F(kb + 3), 1.f, nullptr, q_b,  KP, M, 1000, 0);
      gemm(xv_b, KP, wk_b, KP, Mp, KP, KP, F(kb + 2), 1.f, nullptr, k_b,  KP, M, 1000, 0);
      gemm(xv_b, KP, wv_b, KP, Mp, KP, KP, F(kb + 4), 1.f, nullptr, vT_b, Mp, M, 1000, 1);
      // logits = q.k^T / sqrt(1000), bf16, then in-place softmax
      gemm(q_b, KP, k_b, KP, Mp, Mp, KP, nullptr, 0.031622776601683794f,
           nullptr, attn_b, Mp, M, Mp, 0);
      softmax_rows<<<dim3(M), dim3(256), 0, stream>>>(attn_b, M, Mp);
      // out = attn.v
      gemm(attn_b, Mp, vT_b, Mp, Mp, KP, Mp, nullptr, 1.f, out_f, nullptr, 1000, M, 1000, 0);
      // out1 = LN(out + xv)
      ln_rows<<<dim3(M), dim3(256), 0, stream>>>(out_f, xv_f, F(kb + 6), F(kb + 5),
                                                 out1_f, out1_b, NN, KP);
      // h2 = out1.wd^T + bd ; h3 = LN(h2 + out1)
      gemm(out1_b, KP, wd_b, KP, Mp, KP, KP, F(kb + 0), 1.f, h2_f, nullptr, 1000, M, 1000, 0);
      ln_rows<<<dim3(M), dim3(256), 0, stream>>>(h2_f, out1_f, F(kb + 8), F(kb + 7),
                                                 h3_f, nullptr, NN, KP);
      // scores -> ascending order -> edge conv
      scores_kernel<<<dim3((NN + 255) / 256, KT), dim3(256), 0, stream>>>(
          h3_f, xv_f, F(kb + 10), F(kb + 1), scb, C);
      order20_kernel<<<dim3((NN + 127) / 128), dim3(128), 0, stream>>>(scb, ordb);
      edgeconv_kernel<<<dim3((NN + 127) / 128, O), dim3(128), 0, stream>>>(
          xb, xv_f, ordb, convw, bng, bnb, bnrm, bnrv, xout + (size_t)b * O * NN, C, O);
    }
  };

  edge_block(h0, 10,  64,  34, F(13), F(9),  F(10), F(11), F(12), x1);
  edge_block(x1, 64,  64,  48, F(18), F(14), F(15), F(16), F(17), x2);
  edge_block(x2, 64,  128, 62, F(23), F(19), F(20), F(21), F(22), x3);
  edge_block(x3, 128, 256, 76, F(28), F(24), F(25), F(26), F(27), x4);

  // ================= stage 3: head =================
  ec5max_kernel<<<dim3(1024, BB), dim3(256), 0, stream>>>(
      x1, x2, x3, x4, F(33), F(30), F(29), F(31), F(32), feat);
  dense_kernel<<<dim3((512 + 63) / 64, BB), dim3(64), 0, stream>>>(
      feat, F(90), F(2), F(1), F(3), F(4), fcA, 1024, 512, 1);
  dense_kernel<<<dim3((256 + 63) / 64, BB), dim3(64), 0, stream>>>(
      fcA, F(91), F(6), F(5), F(7), F(8), fcB, 512, 256, 1);
  dense_kernel<<<dim3((40 + 63) / 64, BB), dim3(64), 0, stream>>>(
      fcB, F(92), nullptr, nullptr, nullptr, nullptr, (float*)d_out, 256, 40, 0);
}
